// MultiHeadedAttention_27023934227152
// MI455X (gfx1250) — compile-verified, matrix operand order checked by reading, not run on hardware
//
#include <hip/hip_runtime.h>

typedef __attribute__((ext_vector_type(16))) __bf16 v16bf;
typedef __attribute__((ext_vector_type(8)))  float  v8f;

constexpr int BB = 4, TT = 2048, FF = 1024, HH = 16, DK = 64;

__device__ inline unsigned short f2bf(float f) {
  unsigned u = __float_as_uint(f);
  u += 0x7FFFu + ((u >> 16) & 1u);          // round-to-nearest-even
  return (unsigned short)(u >> 16);
}

union FragU {
  uint4 u4[2];
  unsigned short us[16];
  v16bf v;
};

// ---- CDNA5 async global->LDS copy (ASYNCcnt path) -------------------------
// GLOBAL_LOAD_ASYNC_TO_LDS_B128: vdst = LDS byte offset (addr[31:0] of the
// generic shared pointer), vaddr = 64-bit global address, GV mode (no saddr).
__device__ inline void async_cp_b128(void* lds_ptr, const void* gaddr) {
  unsigned lds_off = (unsigned)(unsigned long long)lds_ptr;
  asm volatile("global_load_async_to_lds_b128 %0, %1, off"
               :: "v"(lds_off), "v"(gaddr) : "memory");
}
__device__ inline void wait_async0() {
  asm volatile("s_wait_asynccnt 0x0" ::: "memory");
}

// 16x32 bf16 fragment load (A or B operand), ISA 7.12.2 layout:
// lane(0..15): row = row16+lane, K chunks [k0+0..7] and [k0+16..23]
// lane(16..31): row = row16+lane-16, K chunks [k0+8..15] and [k0+24..31]
__device__ inline v16bf load_frag_bf16(const unsigned short* p,
                                       int ld, int row16, int k0, int lane) {
  const int hf = lane & 15, hi = lane >> 4;
  const unsigned short* r = p + (size_t)(row16 + hf) * ld + k0 + hi * 8;
  FragU f;
  f.u4[0] = *(const uint4*)(r);
  f.u4[1] = *(const uint4*)(r + 16);
  return f.v;
}

// Same fragment but sourced from fp32 memory, converted in-register.
__device__ inline v16bf load_frag_f32(const float* __restrict__ p,
                                      int ld, int row16, int k0, int lane) {
  const int hf = lane & 15, hi = lane >> 4;
  const float* r = p + (size_t)(row16 + hf) * ld + k0 + hi * 8;
  float4 a0 = *(const float4*)(r);
  float4 a1 = *(const float4*)(r + 4);
  float4 b0 = *(const float4*)(r + 16);
  float4 b1 = *(const float4*)(r + 20);
  FragU f;
  f.us[0]  = f2bf(a0.x); f.us[1]  = f2bf(a0.y); f.us[2]  = f2bf(a0.z); f.us[3]  = f2bf(a0.w);
  f.us[4]  = f2bf(a1.x); f.us[5]  = f2bf(a1.y); f.us[6]  = f2bf(a1.z); f.us[7]  = f2bf(a1.w);
  f.us[8]  = f2bf(b0.x); f.us[9]  = f2bf(b0.y); f.us[10] = f2bf(b0.z); f.us[11] = f2bf(b0.w);
  f.us[12] = f2bf(b1.x); f.us[13] = f2bf(b1.y); f.us[14] = f2bf(b1.z); f.us[15] = f2bf(b1.w);
  return f.v;
}

__device__ inline v8f wmma_bf16(v16bf a, v16bf b, v8f c) {
  return __builtin_amdgcn_wmma_f32_16x16x32_bf16(false, a, false, b, (short)0, c,
                                                 false, false);
}

__device__ inline float rowmax16(float v) {
  #pragma unroll
  for (int m = 1; m < 16; m <<= 1) v = fmaxf(v, __shfl_xor(v, m, 32));
  return v;
}
__device__ inline float rowsum16(float v) {
  #pragma unroll
  for (int m = 1; m < 16; m <<= 1) v += __shfl_xor(v, m, 32);
  return v;
}

__global__ void f32_to_bf16_kernel(const float* __restrict__ src,
                                   unsigned short* __restrict__ dst, int n) {
  int i = blockIdx.x * blockDim.x + threadIdx.x;
  if (i < n) dst[i] = f2bf(src[i]);
}

// C = A[M,K] * W[N,K]^T + bias   (TN GEMM; both operands K-contiguous)
// MODE 0: out = fp32 [M,N]               (final projection -> d_out)
// MODE 1: out = bf16 [B,H,T,DK]          (Q / K head layout)
// MODE 2: out = bf16 [B,H,DK,T]          (V head-transposed for P*V)
// Block: 256 thr = 8 waves; wave w owns rows [m0+32w, +32); block tile 256x64.
// W tile (64n x 32k) is async-staged into double-buffered LDS, shared by all
// 8 waves; per K-step each wave runs 8 WMMAs against 4 LDS B-fragments.
template <int MODE, bool AF32>
__global__ __launch_bounds__(256) void gemm_wmma(const void* __restrict__ Av,
                                                 const unsigned short* __restrict__ W,
                                                 const float* __restrict__ bias,
                                                 void* __restrict__ Ov,
                                                 int M, int N, int K) {
  __shared__ unsigned short wtile[2][64 * 48];   // rows padded 32->48 (96B, 16B-aligned)
  const int tid  = threadIdx.x;
  const int lane = tid & 31;
  const int wave = tid >> 5;
  const int m0 = blockIdx.y * 256 + wave * 32;
  const int n0 = blockIdx.x * 64;

  auto stage = [&](int buf, int k0) {
    const int row = tid >> 2, ch = tid & 3;      // 64 rows x 4 b128 chunks
    const unsigned short* g = W + (size_t)(n0 + row) * K + k0 + ch * 8;
    async_cp_b128(&wtile[buf][row * 48 + ch * 8], g);
  };

  v8f acc[2][4];
  #pragma unroll
  for (int mt = 0; mt < 2; ++mt)
    #pragma unroll
    for (int t = 0; t < 4; ++t) acc[mt][t] = v8f{};

  stage(0, 0);
  for (int k0 = 0; k0 < K; k0 += 32) {
    const int buf = (k0 >> 5) & 1;
    wait_async0();
    __syncthreads();
    if (k0 + 32 < K) stage(buf ^ 1, k0 + 32);

    v16bf af[2];
    #pragma unroll
    for (int mt = 0; mt < 2; ++mt) {
      if constexpr (AF32) {
        af[mt] = load_frag_f32((const float*)Av, K, m0 + mt * 16, k0, lane);
        const float* pf = (const float*)Av +
                          (size_t)(m0 + mt * 16 + (lane & 15)) * K + k0 + 256;
        __builtin_prefetch(pf, 0, 1);            // global_prefetch_b8
      } else {
        af[mt] = load_frag_bf16((const unsigned short*)Av, K, m0 + mt * 16, k0, lane);
      }
    }
    #pragma unroll
    for (int t = 0; t < 4; ++t) {
      v16bf bf = load_frag_bf16(wtile[buf], 48, t * 16, 0, lane);
      acc[0][t] = wmma_bf16(af[0], bf, acc[0][t]);
      acc[1][t] = wmma_bf16(af[1], bf, acc[1][t]);
    }
  }

  const int hf = lane & 15, hi = lane >> 4;
  #pragma unroll
  for (int mt = 0; mt < 2; ++mt) {
    #pragma unroll
    for (int t = 0; t < 4; ++t) {
      const int col = n0 + t * 16 + hf;
      const float bv = bias[col];
      #pragma unroll
      for (int j = 0; j < 8; ++j) {
        const int row = m0 + mt * 16 + j + 8 * hi;  // C/D layout: vgpr j, hi-half +8
        const float val = acc[mt][t][j] + bv;
        if constexpr (MODE == 0) {
          ((float*)Ov)[(size_t)row * N + col] = val;
        } else {
          const int bb = row >> 11, tok = row & (TT - 1);
          const int h = col >> 6,  d   = col & (DK - 1);
          size_t idx;
          if constexpr (MODE == 1)
            idx = ((size_t)(bb * HH + h) * TT + tok) * DK + d;
          else
            idx = ((size_t)(bb * HH + h) * DK + d) * TT + tok;
          ((unsigned short*)Ov)[idx] = f2bf(val);
        }
      }
    }
  }
}

// Flash attention: block = 4 waves, each wave owns a 16-row Q tile.
// K/V tiles for each 32-key step are async-staged into double-buffered LDS
// (shared by all 4 waves). Per step: 4 WMMAs for Q*K^T, online softmax with
// wave32 xor-shuffle row reductions, P routed through per-wave LDS to convert
// C/D layout -> A-fragment layout, 4 WMMAs for P*V against transposed V.
__global__ __launch_bounds__(128) void flash_attn(const unsigned short* __restrict__ Qh,
                                                  const unsigned short* __restrict__ Kh,
                                                  const unsigned short* __restrict__ Vt,
                                                  const int* __restrict__ mask,
                                                  unsigned short* __restrict__ ctx) {
  __shared__ unsigned short Kl[2][32 * 72];   // 32 keys x 64 d, rows padded to 72
  __shared__ unsigned short Vl[2][64 * 48];   // 64 d x 32 keys, rows padded to 48
  __shared__ unsigned short Pl[4][16 * 48];   // per-wave 16x32 P tile

  const int tid  = threadIdx.x;
  const int lane = tid & 31;
  const int wave = tid >> 5;
  const int h = blockIdx.y, b = blockIdx.z;
  const int q0 = blockIdx.x * 64 + wave * 16;
  const int hf = lane & 15, hi = lane >> 4;

  const unsigned short* Qb = Qh + (size_t)(b * HH + h) * TT * DK;
  const unsigned short* Kb = Kh + (size_t)(b * HH + h) * TT * DK;
  const unsigned short* Vb = Vt + (size_t)(b * HH + h) * DK * TT;

  auto stageKV = [&](int buf, int kt) {
    #pragma unroll
    for (int i = 0; i < 2; ++i) {              // K: 32 rows x 8 chunks = 256 xfers
      const int idx = tid + i * 128;
      const int row = idx >> 3, ch = idx & 7;
      async_cp_b128(&Kl[buf][row * 72 + ch * 8],
                    Kb + (size_t)(kt + row) * DK + ch * 8);
    }
    #pragma unroll
    for (int i = 0; i < 2; ++i) {              // V: 64 rows x 4 chunks = 256 xfers
      const int idx = tid + i * 128;
      const int row = idx >> 2, ch = idx & 3;
      async_cp_b128(&Vl[buf][row * 48 + ch * 8],
                    Vb + (size_t)row * TT + kt + ch * 8);
    }
  };

  const v16bf qf0 = load_frag_bf16(Qb, DK, q0, 0, lane);
  const v16bf qf1 = load_frag_bf16(Qb, DK, q0, 32, lane);

  v8f o[4] = {v8f{}, v8f{}, v8f{}, v8f{}};
  float mrow[8], lrow[8];
  #pragma unroll
  for (int j = 0; j < 8; ++j) { mrow[j] = -1e30f; lrow[j] = 0.f; }

  stageKV(0, 0);
  for (int kt = 0; kt < TT; kt += 32) {
    const int buf = (kt >> 5) & 1;
    wait_async0();
    __syncthreads();
    if (kt + 32 < TT) stageKV(buf ^ 1, kt + 32);

    // scores for key groups [kt, kt+16) and [kt+16, kt+32) from LDS K tile
    v8f s[2];
    #pragma unroll
    for (int half = 0; half < 2; ++half) {
      v16bf kf0 = load_frag_bf16(Kl[buf], 72, half * 16, 0, lane);
      v16bf kf1 = load_frag_bf16(Kl[buf], 72, half * 16, 32, lane);
      v8f sv = {};
      sv = wmma_bf16(qf0, kf0, sv);
      sv = wmma_bf16(qf1, kf1, sv);
      s[half] = sv;
    }
    const float mv0 = (mask[b * TT + kt + hf] != 0) ? 1.f : 0.f;
    const float mv1 = (mask[b * TT + kt + 16 + hf] != 0) ? 1.f : 0.f;

    float mnew[8], alpha[8];
    #pragma unroll
    for (int j = 0; j < 8; ++j) {
      const float s0 = s[0][j] * 0.125f;        // 1/sqrt(64)
      const float s1 = s[1][j] * 0.125f;
      s[0][j] = s0; s[1][j] = s1;
      float rm = fmaxf(mv0 != 0.f ? s0 : -1e30f, mv1 != 0.f ? s1 : -1e30f);
      rm = rowmax16(rm);
      mnew[j] = fmaxf(mrow[j], rm);
      alpha[j] = __expf(mrow[j] - mnew[j]);
    }
    #pragma unroll
    for (int j = 0; j < 8; ++j) {
      const float p0 = __expf(s[0][j] - mnew[j]) * mv0;
      const float p1 = __expf(s[1][j] - mnew[j]) * mv1;
      lrow[j] = lrow[j] * alpha[j] + rowsum16(p0 + p1);
      mrow[j] = mnew[j];
      const int row = j + 8 * hi;
      Pl[wave][row * 48 + hf]      = f2bf(p0);
      Pl[wave][row * 48 + 16 + hf] = f2bf(p1);
      #pragma unroll
      for (int t = 0; t < 4; ++t) o[t][j] *= alpha[j];
    }
    __syncthreads();                            // order LDS P write -> A-frag read
    const v16bf pf = load_frag_bf16(Pl[wave], 48, 0, 0, lane);
    #pragma unroll
    for (int t = 0; t < 4; ++t) {
      v16bf vf = load_frag_bf16(Vl[buf], 48, t * 16, 0, lane);
      o[t] = wmma_bf16(pf, vf, o[t]);
    }
  }

  #pragma unroll
  for (int j = 0; j < 8; ++j) {
    const float inv = (lrow[j] > 0.f) ? 1.f / lrow[j] : 0.f;
    const int tok = q0 + j + 8 * hi;
    #pragma unroll
    for (int t = 0; t < 4; ++t) {
      const int col = h * DK + t * 16 + hf;
      ctx[(size_t)(b * TT + tok) * FF + col] = f2bf(o[t][j] * inv);
    }
  }
}

extern "C" void kernel_launch(void* const* d_in, const int* in_sizes, int n_in,
                              void* d_out, int out_size, void* d_ws, size_t ws_size,
                              hipStream_t stream) {
  const float* query = (const float*)d_in[0];
  const float* key   = (const float*)d_in[1];
  const float* value = (const float*)d_in[2];
  const int*   maskp = (const int*)d_in[3];
  const float* Wq = (const float*)d_in[4];  const float* bq = (const float*)d_in[5];
  const float* Wk = (const float*)d_in[6];  const float* bk = (const float*)d_in[7];
  const float* Wv = (const float*)d_in[8];  const float* bv = (const float*)d_in[9];
  const float* Wo = (const float*)d_in[10]; const float* bo = (const float*)d_in[11];

  const size_t nW = (size_t)FF * FF;          // 1M elems per weight
  const size_t nA = (size_t)BB * TT * FF;     // 8M elems per activation tensor
  unsigned short* wq  = (unsigned short*)d_ws;
  unsigned short* wk  = wq + nW;
  unsigned short* wv  = wk + nW;
  unsigned short* wo  = wv + nW;
  unsigned short* Qh  = wo + nW;
  unsigned short* Kh  = Qh + nA;
  unsigned short* Vt  = Kh + nA;
  unsigned short* ctx = Vt + nA;              // total: 8MB + 64MB = 72MB

  const int M = BB * TT, N = FF, K = FF;
  f32_to_bf16_kernel<<<(int)((nW + 255) / 256), 256, 0, stream>>>(Wq, wq, (int)nW);
  f32_to_bf16_kernel<<<(int)((nW + 255) / 256), 256, 0, stream>>>(Wk, wk, (int)nW);
  f32_to_bf16_kernel<<<(int)((nW + 255) / 256), 256, 0, stream>>>(Wv, wv, (int)nW);
  f32_to_bf16_kernel<<<(int)((nW + 255) / 256), 256, 0, stream>>>(Wo, wo, (int)nW);

  dim3 gB(N / 64, M / 256);
  gemm_wmma<1, true><<<gB, 256, 0, stream>>>(query, wq, bq, Qh, M, N, K);
  gemm_wmma<1, true><<<gB, 256, 0, stream>>>(key,   wk, bk, Kh, M, N, K);
  gemm_wmma<2, true><<<gB, 256, 0, stream>>>(value, wv, bv, Vt, M, N, K);

  flash_attn<<<dim3(TT / 64, HH, BB), 128, 0, stream>>>(Qh, Kh, Vt, maskp, ctx);

  gemm_wmma<0, false><<<gB, 256, 0, stream>>>(ctx, wo, bo, d_out, M, N, K);
}
